// SmoothMHA_15960098472040
// MI455X (gfx1250) — matrix operand order repaired and checked by reading, not compiled, not run on hardware
//
#include <hip/hip_runtime.h>

#define D 128
#define NPT 16384
#define BATCH 4
#define KNN 12
#define NHEAD 8

typedef __attribute__((ext_vector_type(16))) __bf16 v16bf;
typedef __attribute__((ext_vector_type(8)))  __bf16 v8bf;
typedef __attribute__((ext_vector_type(8)))  float  v8f;
typedef __attribute__((ext_vector_type(4)))  unsigned int v4u;
typedef __attribute__((ext_vector_type(8)))  int v8i_t;
typedef __attribute__((ext_vector_type(4)))  int v4i_t;

// ---- CDNA5 TDM availability (5-arg ROCm-7.2 form vs 6-arg therock form) ----
#if defined(__has_builtin)
#if __has_builtin(__builtin_amdgcn_tensor_load_to_lds) && \
    __has_builtin(__builtin_amdgcn_s_wait_tensorcnt)
#define USE_TDM 1
#endif
#endif
#ifndef USE_TDM
#define USE_TDM 0
#endif

static __device__ inline v16bf concat8(v8bf a, v8bf b) {
  return __builtin_shufflevector(a, b, 0, 1, 2, 3, 4, 5, 6, 7, 8, 9, 10, 11, 12, 13, 14, 15);
}
static __device__ inline v8bf v8bf_zero() {
  v8bf z;
#pragma unroll
  for (int i = 0; i < 8; ++i) z[i] = (__bf16)0.0f;
  return z;
}
static __device__ inline v16bf bfrag_zero() { return concat8(v8bf_zero(), v8bf_zero()); }
static __device__ inline v8f cfrag_zero() {
  v8f z;
#pragma unroll
  for (int i = 0; i < 8; ++i) z[i] = 0.0f;
  return z;
}
// B operand (contiguous 16 elts per lane): lanes 0-15 K=0..15, lanes 16-31 K=16..31
static __device__ inline v16bf bfrag_load(const __bf16* p) { return *(const v16bf*)p; }
// A operand (contiguous 16 elts per lane): lanes 0-15 K=0..15, lanes 16-31 K=16..31
static __device__ inline v16bf afrag_load(const __bf16* p, int hi) {
  v8bf a = *(const v8bf*)(p + hi * 16);
  v8bf b = *(const v8bf*)(p + hi * 16 + 8);
  return concat8(a, b);
}
// A operand with only K=0..15 valid (lanes 0-15 K=0..15; lanes 16-31 = K16..31 zero-padded)
static __device__ inline v16bf afrag_load_k16(const __bf16* p, int hi) {
  if (hi) return bfrag_zero();
  return concat8(*(const v8bf*)p, *(const v8bf*)(p + 8));
}
static __device__ inline v8f wmma_bf16(v16bf a, v16bf b, v8f c) {
  return __builtin_amdgcn_wmma_f32_16x16x32_bf16(false, a, false, b, (short)0, c, false, false);
}
static __device__ inline unsigned int pack2bf(float a, float b) {
  unsigned short lo = __builtin_bit_cast(unsigned short, (__bf16)a);
  unsigned short hi = __builtin_bit_cast(unsigned short, (__bf16)b);
  return (unsigned int)lo | ((unsigned int)hi << 16);
}

#if USE_TDM
// DMA one contiguous 32KB key tile (4096 x 8-byte units) global -> LDS via TDM.
// D# per CDNA5 ISA ch.8: group0 = {count, lds_addr, global_addr, type=2},
// group1 = {data_size=3(8B), tensor_dim0=4096, tensor_dim1=1, tile_dim0=4096,
//           tile_dim1=1, tensor_dim0_stride=4096}; groups 2/3 zero (<=2D).
static __device__ inline void tdm_load_tile(const __bf16* gsrc, unsigned lds_byte_off) {
  unsigned long long ga = (unsigned long long)(size_t)gsrc;
  v4u g0;
  g0[0] = 1u;                                              // count=1, user mode
  g0[1] = lds_byte_off;                                    // lds_addr
  g0[2] = (unsigned)(ga & 0xFFFFFFFFu);                    // global_addr[31:0]
  g0[3] = (unsigned)((ga >> 32) & 0x1FFFFFFu) | (2u << 30);// global_addr[56:32] | type=2
  v8i_t g1;
  g1[0] = (int)(3u << 16);        // workgroup_mask=0 | data_size=3 (8B)
  g1[1] = (int)(4096u << 16);     // bits[47:32]=0 | tensor_dim0[15:0]=4096 in bits[63:48]
  g1[2] = (int)(1u << 16);        // tensor_dim0[31:16]=0 | tensor_dim1[15:0]=1
  g1[3] = (int)(4096u << 16);     // tensor_dim1[31:16]=0 | tile_dim0=4096
  g1[4] = 1;                      // tile_dim1=1 | tile_dim2=0
  g1[5] = 4096;                   // tensor_dim0_stride[31:0]
  g1[6] = 0;                      // stride0 hi | stride1 lo
  g1[7] = 0;                      // stride1 hi
  v4i_t z4;
#pragma unroll
  for (int i = 0; i < 4; ++i) z4[i] = 0;
#if __has_include(<hip/amd_detail/amd_gfx1250_TDM.h>)
  v8i_t z8;
#pragma unroll
  for (int i = 0; i < 8; ++i) z8[i] = 0;
  __builtin_amdgcn_tensor_load_to_lds(g0, g1, z4, z4, z8, 0);   // 6-arg toolchain
#else
  __builtin_amdgcn_tensor_load_to_lds(g0, g1, z4, z4, 0);       // 5-arg toolchain
#endif
}
#endif

// ---------------- Stage 0a: x -> bf16 copy + squared norms ----------------
__global__ __launch_bounds__(256) void prep_x_kernel(const float* __restrict__ x,
                                                     __bf16* __restrict__ xbf,
                                                     float* __restrict__ sq) {
  const int lane = threadIdx.x & 31;
  const int w = threadIdx.x >> 5;
  const size_t row = (size_t)blockIdx.x * 8 + w;
  const float4 v = ((const float4*)(x + row * D))[lane];
  uint2 pk;
  pk.x = pack2bf(v.x, v.y);
  pk.y = pack2bf(v.z, v.w);
  ((uint2*)(xbf + row * D))[lane] = pk;
  float s = v.x * v.x + v.y * v.y + v.z * v.z + v.w * v.w;
#pragma unroll
  for (int st = 1; st < 32; st <<= 1) s += __shfl_xor(s, st, 32);
  if (lane == 0) sq[row] = s;
}

// ---------------- Stage 0b: weights -> bf16 (Wq,Wk,Wv,Wo row-major) ----------------
__global__ __launch_bounds__(256) void prep_w_kernel(const float* __restrict__ in_proj,
                                                     const float* __restrict__ out_proj,
                                                     __bf16* __restrict__ wbf) {
  const int i = blockIdx.x * blockDim.x + threadIdx.x;
  float f = (i < 3 * D * D) ? in_proj[i] : out_proj[i - 3 * D * D];
  wbf[i] = (__bf16)f;
}

// ---------------- Stage 1: fused GEMM distance + top-12 KNN ----------------
// Grid: BATCH * (NPT/128) blocks of 256 threads (8 waves).
// TDM double-buffers the 32KB key tile into LDS while WMMA + top-k run on the
// previous tile. Ranking key: e = |k|^2 - 2*q.k (same order as true sq dist).
__global__ __launch_bounds__(256) void knn_kernel(const __bf16* __restrict__ xbf,
                                                  const float* __restrict__ sq,
                                                  int* __restrict__ gidx) {
  extern __shared__ char smem[];
  __bf16* keybuf = (__bf16*)smem;                 // 2 x 128x128 bf16 (2 x 32KB)
  float* dist = (float*)(smem + 65536);           // 128x128 f32 (64KB)
  float* cand_d = dist;                           // overlay (after last sync)
  int* cand_i = (int*)(smem + 65536 + 16384);     // overlay within dist

  const int tid = threadIdx.x;
  const int lane = tid & 31, w = tid >> 5;
  const int hi = lane >> 4, l15 = lane & 15;
  const int b = blockIdx.x >> 7;
  const int qbase = (blockIdx.x & 127) * 128;
  const __bf16* xb = xbf + (size_t)b * NPT * D;
  const float* sqb = sq + (size_t)b * NPT;

  // query A fragments live in registers for the whole kernel
  v16bf afr[4];
  {
    const __bf16* qrow = xb + (size_t)(qbase + w * 16 + l15) * D;
#pragma unroll
    for (int ks = 0; ks < 4; ++ks) afr[ks] = afrag_load(qrow + ks * 32, hi);
  }

  const int r = tid >> 1, seg = tid & 1;  // 2 scan threads per query row
  float bd[KNN];
  int bi[KNN];
#pragma unroll
  for (int j = 0; j < KNN; ++j) { bd[j] = 3.0e38f; bi[j] = 0; }

#if USE_TDM
  const unsigned key_lds0 = (unsigned)(unsigned long long)(size_t)keybuf;
  if (w == 0) tdm_load_tile(xb, key_lds0);  // prime tile 0 into buffer 0
#endif

  for (int kt = 0; kt < 128; ++kt) {
    const int keybase = kt * 128;
    __bf16* keyt = keybuf + (size_t)(kt & 1) * (128 * D);
#if USE_TDM
    if (w == 0) __builtin_amdgcn_s_wait_tensorcnt(0);
    __syncthreads();  // publish DMA'd tile to all waves
    if (w == 0 && kt + 1 < 128)
      tdm_load_tile(xb + (size_t)(keybase + 128) * D,
                    key_lds0 + (unsigned)(((kt + 1) & 1) * 32768));
#else
    {  // manual staging fallback
      const uint4* s4 = (const uint4*)(xb + (size_t)(keybase + (tid >> 1)) * D + (tid & 1) * 64);
      uint4* d4 = (uint4*)(keyt + (tid >> 1) * D + (tid & 1) * 64);
#pragma unroll
      for (int i = 0; i < 8; ++i) d4[i] = s4[i];
    }
    __syncthreads();
#endif
    // wave w: 16x128 strip of the distance tile; 8 col subtiles x 4 k-steps
#pragma unroll
    for (int ct = 0; ct < 8; ++ct) {
      v8f acc = cfrag_zero();
#pragma unroll
      for (int ks = 0; ks < 4; ++ks) {
        v16bf bfr = bfrag_load(keyt + (ct * 16 + l15) * D + ks * 32 + hi * 16);
        acc = wmma_bf16(afr[ks], bfr, acc);
      }
      const float kn = sqb[keybase + ct * 16 + l15];
#pragma unroll
      for (int c = 0; c < 8; ++c) {
        const int m = w * 16 + hi * 8 + c;  // C layout: lanes0-15 -> M=c, lanes16-31 -> M=c+8
        dist[m * 128 + ct * 16 + l15] = kn - 2.0f * acc[c];
      }
    }
    __syncthreads();
    {  // per-row top-12 maintenance (sorted insertion, register-resident)
      const float* dr = dist + r * 128 + seg * 64;
      for (int j = 0; j < 64; ++j) {
        const float v = dr[j];
        if (v < bd[KNN - 1]) {
          bd[KNN - 1] = v;
          bi[KNN - 1] = keybase + seg * 64 + j;
#pragma unroll
          for (int q = KNN - 1; q > 0; --q) {
            if (bd[q] < bd[q - 1]) {
              float td = bd[q]; bd[q] = bd[q - 1]; bd[q - 1] = td;
              int ti = bi[q]; bi[q] = bi[q - 1]; bi[q - 1] = ti;
            }
          }
        }
      }
    }
    __syncthreads();
  }
  // merge the two sorted segment lists per row -> final 12
#pragma unroll
  for (int j = 0; j < KNN; ++j) { cand_d[tid * KNN + j] = bd[j]; cand_i[tid * KNN + j] = bi[j]; }
  __syncthreads();
  if (seg == 0) {
    const float* d0 = cand_d + (2 * r) * KNN;
    const float* d1 = cand_d + (2 * r + 1) * KNN;
    const int* i0 = cand_i + (2 * r) * KNN;
    const int* i1 = cand_i + (2 * r + 1) * KNN;
    int a = 0, c = 0;
    int* outp = gidx + ((size_t)b * NPT + qbase + r) * KNN;
    for (int k = 0; k < KNN; ++k) {
      if (d0[a] <= d1[c]) { outp[k] = i0[a]; ++a; }
      else { outp[k] = i1[c]; ++c; }
    }
  }
}

// ---------------- Stage 2: gather + neighborhood MHA + sigmoid pooling ----------------
// One wave per point (12 neighbors padded to 16 rows). dh=16 => one WMMA per head
// for scores and for a@v. Per-wave LDS scratch for layout transposes.
#define WAVE_SCRATCH 16960
__global__ __launch_bounds__(256) void mha_kernel(const __bf16* __restrict__ xbf,
                                                  const __bf16* __restrict__ wbf,
                                                  const int* __restrict__ gidx,
                                                  const float* __restrict__ w_weight,
                                                  const float* __restrict__ w_bias,
                                                  float* __restrict__ out) {
  extern __shared__ char smem[];
  const int tid = threadIdx.x;
  const int lane = tid & 31, w = tid >> 5;
  const int hi = lane >> 4, l15 = lane & 15;
  char* wsb = smem + w * WAVE_SCRATCH;
  __bf16* qs = (__bf16*)wsb;              // 16x128 row-major
  __bf16* kscr = (__bf16*)(wsb + 4096);   // 16x128 row-major
  __bf16* vT = (__bf16*)(wsb + 8192);     // 128x16 (v transposed)
  __bf16* os = (__bf16*)(wsb + 12288);    // 16x128 row-major
  __bf16* ascr = (__bf16*)(wsb + 16384);  // 16x16
  float* xw = (float*)(wsb + 16896);      // 16 pooling weights

  const __bf16* wo = wbf + 3 * D * D;
  const float wvec = (l15 < KNN) ? w_weight[l15] : 0.0f;
  const float bias = w_bias[0];

  for (int it = 0; it < 8; ++it) {
    const int p = blockIdx.x * 64 + w * 8 + it;
    const int b = p >> 14;  // NPT = 16384

    // gather neighborhood rows as WMMA A fragments (rows 12..15 zero)
    v16bf ga[4];
    if (l15 < KNN) {
      const int nb = gidx[(size_t)p * KNN + l15];
      const __bf16* grow = xbf + ((size_t)b * NPT + nb) * D;
#pragma unroll
      for (int kk = 0; kk < 4; ++kk) ga[kk] = afrag_load(grow + kk * 32, hi);
    } else {
#pragma unroll
      for (int kk = 0; kk < 4; ++kk) ga[kk] = bfrag_zero();
    }

    // q/k/v projections: [16x128] = g @ W^T ; W rows are contiguous => direct B frags
#pragma unroll
    for (int proj = 0; proj < 3; ++proj) {
      const __bf16* W = wbf + proj * D * D;
#pragma unroll
      for (int ct = 0; ct < 8; ++ct) {
        v8f acc = cfrag_zero();
#pragma unroll
        for (int kk = 0; kk < 4; ++kk) {
          v16bf bfr = bfrag_load(W + (ct * 16 + l15) * D + kk * 32 + hi * 16);
          acc = wmma_bf16(ga[kk], bfr, acc);
        }
        if (proj < 2) {
          __bf16* dst = (proj == 0) ? qs : kscr;
#pragma unroll
          for (int c = 0; c < 8; ++c)
            dst[(hi * 8 + c) * D + ct * 16 + l15] = (__bf16)acc[c];
        } else {  // v stored transposed: column e is contiguous -> packed 16B store
          uint4 pk;
          pk.x = pack2bf(acc[0], acc[1]);
          pk.y = pack2bf(acc[2], acc[3]);
          pk.z = pack2bf(acc[4], acc[5]);
          pk.w = pack2bf(acc[6], acc[7]);
          *(uint4*)(vT + (ct * 16 + l15) * 16 + hi * 8) = pk;
        }
      }
    }

    // per-head attention; accumulate head-mean attn weights in aw[]
    float aw[8];
#pragma unroll
    for (int c = 0; c < 8; ++c) aw[c] = 0.0f;
#pragma unroll 1
    for (int h = 0; h < NHEAD; ++h) {
      v16bf qa = afrag_load_k16(qs + l15 * D + h * 16, hi);
      v16bf kb = hi ? bfrag_zero() : bfrag_load(kscr + l15 * D + h * 16);
      v8f s = wmma_bf16(qa, kb, cfrag_zero());
      float ev[8], red[8];
#pragma unroll
      for (int c = 0; c < 8; ++c) {
        float t = 0.25f * s[c];  // 1/sqrt(dh)
        if (l15 >= KNN) t = -1.0e30f;
        ev[c] = t;
        red[c] = t;
      }
#pragma unroll
      for (int st = 1; st < 16; st <<= 1)
#pragma unroll
        for (int c = 0; c < 8; ++c) red[c] = fmaxf(red[c], __shfl_xor(red[c], st, 16));
#pragma unroll
      for (int c = 0; c < 8; ++c) { ev[c] = __expf(ev[c] - red[c]); red[c] = ev[c]; }
#pragma unroll
      for (int st = 1; st < 16; st <<= 1)
#pragma unroll
        for (int c = 0; c < 8; ++c) red[c] += __shfl_xor(red[c], st, 16);
#pragma unroll
      for (int c = 0; c < 8; ++c) {
        const float a = ev[c] / red[c];
        aw[c] += a;
        ascr[(hi * 8 + c) * 16 + l15] = (__bf16)a;
      }
      // o_h = a @ v_h : A = a rows (K=12 padded), B = vT columns
      v16bf aa = afrag_load_k16(ascr + l15 * 16, hi);
      v16bf vb = hi ? bfrag_zero() : bfrag_load(vT + (h * 16 + l15) * 16);
      v8f o = wmma_bf16(aa, vb, cfrag_zero());
#pragma unroll
      for (int c = 0; c < 8; ++c)
        os[(hi * 8 + c) * D + h * 16 + l15] = (__bf16)o[c];
    }

    // sigmoid pooling weights from head-averaged attention
    float wp[8];
    {
      float tv[8];
#pragma unroll
      for (int c = 0; c < 8; ++c) tv[c] = aw[c] * 0.125f * wvec;
#pragma unroll
      for (int st = 1; st < 16; st <<= 1)
#pragma unroll
        for (int c = 0; c < 8; ++c) tv[c] += __shfl_xor(tv[c], st, 16);
#pragma unroll
      for (int c = 0; c < 8; ++c) {
        const int m = hi * 8 + c;
        const float xs = (m < KNN) ? (1.0f / (1.0f + __expf(-(tv[c] + bias)))) : 0.0f;
        if (l15 == c) xw[m] = xs;
      }
      float ssum = 0.0f;
#pragma unroll
      for (int m = 0; m < KNN; ++m) ssum += xw[m];
      const float inv = 1.0f / ssum;
#pragma unroll
      for (int c = 0; c < 8; ++c) wp[c] = xw[hi * 8 + c] * inv;
    }

    // attn_out = o @ Wo^T, fused with weighted pooling over rows -> out[p, :]
    v16bf oa[4];
#pragma unroll
    for (int kk = 0; kk < 4; ++kk) oa[kk] = afrag_load(os + l15 * D + kk * 32, hi);
#pragma unroll
    for (int ct = 0; ct < 8; ++ct) {
      v8f acc = cfrag_zero();
#pragma unroll
      for (int kk = 0; kk < 4; ++kk) {
        v16bf bfr = bfrag_load(wo + (ct * 16 + l15) * D + kk * 32 + hi * 16);
        acc = wmma_bf16(oa[kk], bfr, acc);
      }
      float rsum = 0.0f;
#pragma unroll
      for (int c = 0; c < 8; ++c) rsum += wp[c] * acc[c];
      rsum += __shfl_xor(rsum, 16, 32);  // combine M=0-7 half with M=8-15 half
      if (!hi) out[(size_t)p * D + ct * 16 + l15] = rsum;
    }
  }
}

extern "C" void kernel_launch(void* const* d_in, const int* in_sizes, int n_in,
                              void* d_out, int out_size, void* d_ws, size_t ws_size,
                              hipStream_t stream) {
  (void)in_sizes; (void)n_in; (void)out_size; (void)ws_size;
  const float* x = (const float*)d_in[0];
  const float* in_proj = (const float*)d_in[1];
  const float* out_proj = (const float*)d_in[2];
  const float* w_weight = (const float*)d_in[3];
  const float* w_bias = (const float*)d_in[4];
  float* out = (float*)d_out;

  char* ws = (char*)d_ws;
  __bf16* xbf = (__bf16*)ws;
  size_t off = (size_t)BATCH * NPT * D * 2;
  float* sq = (float*)(ws + off);
  off += (size_t)BATCH * NPT * 4;
  int* gidx = (int*)(ws + off);
  off += (size_t)BATCH * NPT * KNN * 4;
  __bf16* wbf = (__bf16*)(ws + off);

  prep_x_kernel<<<BATCH * NPT / 8, 256, 0, stream>>>(x, xbf, sq);
  prep_w_kernel<<<(4 * D * D) / 256, 256, 0, stream>>>(in_proj, out_proj, wbf);

  const int shm1 = 65536 + 65536;  // 2 key buffers + distance tile
  (void)hipFuncSetAttribute((const void*)knn_kernel,
                            hipFuncAttributeMaxDynamicSharedMemorySize, shm1);
  knn_kernel<<<BATCH * (NPT / 128), 256, shm1, stream>>>(xbf, sq, gidx);

  const int shm2 = 8 * WAVE_SCRATCH;
  (void)hipFuncSetAttribute((const void*)mha_kernel,
                            hipFuncAttributeMaxDynamicSharedMemorySize, shm2);
  mha_kernel<<<(BATCH * NPT) / 64, 256, shm2, stream>>>(xbf, wbf, gidx, w_weight, w_bias, out);
}
